// HydraNet_13846974562836
// MI455X (gfx1250) — matrix operand order repaired and checked by reading, not compiled, not run on hardware
//
#include <hip/hip_runtime.h>

typedef _Float16 h16;
typedef _Float16 v16h __attribute__((ext_vector_type(16)));
typedef float    v8f  __attribute__((ext_vector_type(8)));

#define NPAR 93
struct Params { const float* a[NPAR]; };

// d_in layout: 0=y_p1, 1=y_p2, then per player p (40 arrays), per FE f (10 arrays):
//   ln1_g, ln1_b, W1, b1, W2, b2, ln2_g, ln2_b, Ws, bs
#define FEBASE(p, f) (2 + (p)*40 + (f)*10)
// 82=Wqkv[192,64] 83=bqkv 84=Wo[64,64] 85=bo 86=aw[4] 87=clsW1[128,64] 88=clsb1
// 89=cls ln_g 90=cls ln_b 91=clsW2[64,1] 92=clsb2

// ---------- packed f16 weight layout (element = half) ----------
// per player p (stride 24576): w1blk[256][16] @0 | wsblk[256][16] @4096 | w2[4][64][64] @8192
// wqkv[192][64] @49152 | wo[64][64] @61440 | clsw1[64][128] @65536  -> 73728 halves total
// ---------- misc f32 layout ----------
// per player (stride 1312): ln1g[16]@0 ln1b[16]@16 b1cat[256]@32 b2cat[256]@288
//   ln2g[256]@544 ln2b[256]@800 bscat[256]@1056
// bqkv[192]@2624 bo[64]@2816 awsoft[4]@2880 clsb1[64]@2884 clslng[64]@2948
// clslnb[64]@3012 clsw2[64]@3076 clsb2[1]@3140   (3141 floats)

__device__ __forceinline__ int fe_of(int j) { return j < 6 ? 0 : (j < 8 ? 1 : (j < 15 ? 2 : 3)); }

__device__ __forceinline__ v8f zero8() {
  v8f z;
#pragma unroll
  for (int i = 0; i < 8; ++i) z[i] = 0.f;
  return z;
}

__device__ __forceinline__ v8f wmma16(v16h a, v16h b, v8f c) {
  return __builtin_amdgcn_wmma_f32_16x16x32_f16(false, a, false, b, (short)0, c, false, false);
}

// A fragment (MxK, row-major in LDS): lane-half interleave per ISA 16-bit A table.
__device__ __forceinline__ v16h ldsA(const h16* base, int stride, int row, int kbase, int koff) {
  v16h a;
  const h16* p = base + row * stride + kbase + koff;
#pragma unroll
  for (int i = 0; i < 8; ++i) a[i] = p[i];
#pragma unroll
  for (int i = 0; i < 8; ++i) a[8 + i] = p[16 + i];
  return a;
}
// A fragment with only K=0..15 valid (K 16..31 zero-padded on the A side)
__device__ __forceinline__ v16h ldsA16(const h16* base, int stride, int row, int koff) {
  v16h a;
  const h16* p = base + row * stride + koff;
#pragma unroll
  for (int i = 0; i < 8; ++i) a[i] = p[i];
#pragma unroll
  for (int i = 0; i < 8; ++i) a[8 + i] = (h16)0.f;
  return a;
}
// B fragment (KxN stored col-major [n][K]): table order, lane-half h gets K 8h..8h+7 and 16+8h..16+8h+7
__device__ __forceinline__ v16h ldsB(const h16* base, int kstride, int n, int kbase, int ksel) {
  v16h b;
  const h16* p = base + n * kstride + kbase + ksel / 2;
#pragma unroll
  for (int i = 0; i < 8; ++i) b[i] = p[i];
#pragma unroll
  for (int i = 0; i < 8; ++i) b[8 + i] = p[16 + i];
  return b;
}
// B fragment for K=16 matrices ([n][16] packed). Elements 8-15 of each lane (which feed K 16..31)
// simply re-read the same finite weight values; the A fragment's K 16..31 elements are
// zero, so those products contribute exactly 0 (finite*0 == 0, no NaN possible).
// This avoids any EXEC-divergent zero-fill in the inner loops.
__device__ __forceinline__ v16h ldsB16(const h16* base, int n) {
  v16h b;
  const h16* p = base + n * 16 + ((threadIdx.x >> 4) & 1) * 8;
#pragma unroll
  for (int i = 0; i < 16; ++i) b[i] = p[i & 7];
  return b;
}

// =================== weight pack / transpose kernel ===================
__global__ void __launch_bounds__(256) pack_weights(Params P, h16* __restrict__ Wg,
                                                    float* __restrict__ Mg) {
  const int tid = blockIdx.x * blockDim.x + threadIdx.x;
  const int nth = gridDim.x * blockDim.x;
  const int dn[4] = {6, 2, 7, 1};
  const int ro[4] = {0, 6, 8, 15};

  // block-diagonal first-layer + shortcut weights: [n=256][k=16]
  for (int i = tid; i < 2 * 256 * 16; i += nth) {
    int p = i >> 12, n = (i >> 4) & 255, k = i & 15;
    int f = n >> 6, col = n & 63;
    int kl = k - ro[f];
    float v1 = 0.f, vs = 0.f;
    if (kl >= 0 && kl < dn[f]) {
      v1 = P.a[FEBASE(p, f) + 2][kl * 64 + col];  // W1 (din,64)
      vs = P.a[FEBASE(p, f) + 8][kl * 64 + col];  // Ws (din,64)
    }
    Wg[p * 24576 + n * 16 + k] = (h16)v1;
    Wg[p * 24576 + 4096 + n * 16 + k] = (h16)vs;
  }
  // W2 transposed to [n][k]
  for (int i = tid; i < 2 * 4 * 4096; i += nth) {
    int p = i >> 14, f = (i >> 12) & 3, n = (i >> 6) & 63, k = i & 63;
    Wg[p * 24576 + 8192 + f * 4096 + n * 64 + k] = (h16)P.a[FEBASE(p, f) + 4][k * 64 + n];
  }
  // Wqkv / Wo already [out][in] == [n][k]
  for (int i = tid; i < 192 * 64; i += nth) Wg[49152 + i] = (h16)P.a[82][i];
  for (int i = tid; i < 64 * 64; i += nth) Wg[61440 + i] = (h16)P.a[84][i];
  // cls W1 (128,64) transposed to [n=64][k=128]
  for (int i = tid; i < 64 * 128; i += nth) {
    int n = i >> 7, k = i & 127;
    Wg[65536 + i] = (h16)P.a[87][k * 64 + n];
  }
  // misc f32
  for (int i = tid; i < 2 * 1312; i += nth) {
    int p = i / 1312, j = i % 1312;
    float v;
    if (j < 16)        { int f = fe_of(j);                 v = P.a[FEBASE(p, f) + 0][j - ro[f]]; }
    else if (j < 32)   { int jj = j - 16; int f = fe_of(jj); v = P.a[FEBASE(p, f) + 1][jj - ro[f]]; }
    else if (j < 288)  { int jj = j - 32;   v = P.a[FEBASE(p, jj >> 6) + 3][jj & 63]; }
    else if (j < 544)  { int jj = j - 288;  v = P.a[FEBASE(p, jj >> 6) + 5][jj & 63]; }
    else if (j < 800)  { int jj = j - 544;  v = P.a[FEBASE(p, jj >> 6) + 6][jj & 63]; }
    else if (j < 1056) { int jj = j - 800;  v = P.a[FEBASE(p, jj >> 6) + 7][jj & 63]; }
    else               { int jj = j - 1056; v = P.a[FEBASE(p, jj >> 6) + 9][jj & 63]; }
    Mg[p * 1312 + j] = v;
  }
  for (int i = tid; i < 192; i += nth) Mg[2624 + i] = P.a[83][i];
  for (int i = tid; i < 64; i += nth) Mg[2816 + i] = P.a[85][i];
  for (int i = tid; i < 4; i += nth) {  // softmax(aw)
    float m = P.a[86][0];
    for (int j = 1; j < 4; ++j) m = fmaxf(m, P.a[86][j]);
    float e[4]; float s = 0.f;
    for (int j = 0; j < 4; ++j) { e[j] = __expf(P.a[86][j] - m); s += e[j]; }
    Mg[2880 + i] = e[i] / s;
  }
  for (int i = tid; i < 64; i += nth) Mg[2884 + i] = P.a[88][i];
  for (int i = tid; i < 64; i += nth) Mg[2948 + i] = P.a[89][i];
  for (int i = tid; i < 64; i += nth) Mg[3012 + i] = P.a[90][i];
  for (int i = tid; i < 64; i += nth) Mg[3076 + i] = P.a[91][i];
  if (tid == 0) Mg[3140] = P.a[92][0];
}

// =================== fused forward kernel ===================
// block = 128 threads = 4 waves; each wave owns 16 samples.
__global__ void __launch_bounds__(128) hydranet_fused(
    const float* __restrict__ y1, const float* __restrict__ y2,
    const h16* __restrict__ wf16g, const float* __restrict__ miscg,
    float* __restrict__ out, float* __restrict__ partials) {
  extern __shared__ char smem[];
  h16* W = (h16*)smem;                               // 73728 halves
  float* M = (float*)(smem + 147456);                // 3141 used (+pad)
  float* VERS = M + 3200;                            // 128 floats
  h16* wavemem = (h16*)(smem + 147456 + 13312);

  const int lane = threadIdx.x & 31;
  const int wv = threadIdx.x >> 5;
  const int r = lane & 15;
  const int hf = lane >> 4;
  const int koff = hf * 8;
  const int ksel = hf * 16;

  // stage weights into LDS
  {
    const int4* s = (const int4*)wf16g;
    int4* d = (int4*)W;
    for (int i = threadIdx.x; i < 9216; i += blockDim.x) d[i] = s[i];
    for (int i = threadIdx.x; i < 3141; i += blockDim.x) M[i] = miscg[i];
  }
  __syncthreads();

  h16* XLN = wavemem + wv * 17408;  // 32x16 LN'd inputs (rows 0-15 p1, 16-31 p2)
  h16* XRW = XLN + 512;             // 32x16 raw inputs
  h16* TOK = XRW + 512;             // 128x64 tokens (sample*8 + token)
  h16* SCR = TOK + 8192;            // 6144 halves scratch
  h16* FB = SCR + 6144;             // 16x128 fused features (doubles as f32 LN-exchange)
  float* EXF = (float*)FB;          // per-wave partial-reduction exchange (dead phases only)

  // ---- stage 1: load, per-sub-block LayerNorm, versus term ----
  const int base = blockIdx.x * 64 + wv * 16;
  const float* srcrow = (hf ? y2 : y1) + (size_t)(base + r) * 16;
  float x[16];
  {
    const float4* s4 = (const float4*)srcrow;
#pragma unroll
    for (int i = 0; i < 4; ++i) {
      float4 t = s4[i];
      x[4 * i] = t.x; x[4 * i + 1] = t.y; x[4 * i + 2] = t.z; x[4 * i + 3] = t.w;
    }
  }
  {
    const int ro[4] = {0, 6, 8, 15}, dn[4] = {6, 2, 7, 1};
    const float* g = M + hf * 1312;
    const float* bt = M + hf * 1312 + 16;
#pragma unroll
    for (int b = 0; b < 4; ++b) {
      float mu = 0.f;
      for (int i = 0; i < dn[b]; ++i) mu += x[ro[b] + i];
      mu /= (float)dn[b];
      float var = 0.f;
      for (int i = 0; i < dn[b]; ++i) { float d0 = x[ro[b] + i] - mu; var += d0 * d0; }
      float inv = rsqrtf(var / (float)dn[b] + 1e-5f);
      for (int i = 0; i < dn[b]; ++i) {
        int idx = ro[b] + i;
        XLN[lane * 16 + idx] = (h16)((x[idx] - mu) * inv * g[idx] + bt[idx]);
      }
    }
#pragma unroll
    for (int i = 0; i < 16; ++i) XRW[lane * 16 + i] = (h16)x[i];
  }
  float vv = 0.f;
  if (hf == 0) {
    const float* s2 = y2 + (size_t)(base + r) * 16;
    float d0 = 0.f, q1 = 0.f, q2 = 0.f;
#pragma unroll
    for (int i = 0; i < 16; ++i) { float b2 = s2[i]; d0 += x[i] * b2; q1 += x[i] * x[i]; q2 += b2 * b2; }
    float c = d0 / (fmaxf(sqrtf(q1), 1e-12f) * fmaxf(sqrtf(q2), 1e-12f));
    vv = fmaxf(1.f + c, 0.f);
  }
  VERS[threadIdx.x] = vv;
  __syncthreads();
  if (threadIdx.x == 0) {
    float s = 0.f;
    for (int i = 0; i < 128; ++i) s += VERS[i];
    partials[blockIdx.x] = s;
  }

  // ---- stage 2: FeatureEnhancement for both players (WMMA) ----
  h16* H1 = SCR;         // 16x256 relu(x@W1blk+b1)
  h16* H2T = SCR + 4096; // 16x64
  h16* SCT = SCR + 5120; // 16x64
  for (int p = 0; p < 2; ++p) {
    const h16* W1B = W + p * 24576;
    const h16* WSB = W1B + 4096;
    const h16* W2B = W1B + 8192;
    v16h Aln = ldsA16(XLN + p * 256, 16, r, koff);
    v16h Arw = ldsA16(XRW + p * 256, 16, r, koff);
    const float* b1c = M + p * 1312 + 32;
#pragma unroll 4
    for (int nt = 0; nt < 16; ++nt) {
      v8f c = zero8();
      c = wmma16(Aln, ldsB16(W1B, nt * 16 + r), c);
      int col = nt * 16 + r;
      float bv = b1c[col];
#pragma unroll
      for (int v = 0; v < 8; ++v) H1[(v + hf * 8) * 256 + col] = (h16)fmaxf(c[v] + bv, 0.f);
    }
    __syncthreads();
    for (int f = 0; f < 4; ++f) {
      v16h a0 = ldsA(H1, 256, r, f * 64, koff);
      v16h a1 = ldsA(H1, 256, r, f * 64 + 32, koff);
      const float* b2c = M + p * 1312 + 288 + f * 64;
      const float* bsc = M + p * 1312 + 1056 + f * 64;
#pragma unroll
      for (int nt = 0; nt < 4; ++nt) {
        v8f c = zero8();
        c = wmma16(a0, ldsB(W2B + f * 4096, 64, nt * 16 + r, 0, ksel), c);
        c = wmma16(a1, ldsB(W2B + f * 4096, 64, nt * 16 + r, 32, ksel), c);
        v8f s = zero8();
        s = wmma16(Arw, ldsB16(WSB, f * 64 + nt * 16 + r), s);
        int col = nt * 16 + r;
        float b2v = b2c[col], bsv = bsc[col];
#pragma unroll
        for (int v = 0; v < 8; ++v) {
          H2T[(v + hf * 8) * 64 + col] = (h16)(c[v] + b2v);
          SCT[(v + hf * 8) * 64 + col] = (h16)(s[v] + bsv);
        }
      }
      __syncthreads();
      {  // row-wise LN + shortcut -> token, all 32 lanes (lane-half owns 32 cols of row r)
        const float* l2g = M + p * 1312 + 544 + f * 64;
        const float* l2b = M + p * 1312 + 800 + f * 64;
        float s = 0.f, q = 0.f;
#pragma unroll
        for (int cc = 0; cc < 32; ++cc) {
          float v0 = (float)H2T[r * 64 + hf * 32 + cc];
          s += v0; q += v0 * v0;
        }
        EXF[lane] = s; EXF[64 + lane] = q;
        __syncthreads();
        float st = EXF[r] + EXF[r + 16];
        float qt = EXF[64 + r] + EXF[64 + r + 16];
        float mu = st * (1.f / 64.f);
        float var = qt * (1.f / 64.f) - mu * mu;
        float inv = rsqrtf(var + 1e-5f);
        h16* trow = TOK + (r * 8 + p * 4 + f) * 64;
#pragma unroll
        for (int cc = 0; cc < 32; ++cc) {
          int idx = hf * 32 + cc;
          float hv = ((float)H2T[r * 64 + idx] - mu) * inv * l2g[idx] + l2b[idx]
                     + (float)SCT[r * 64 + idx];
          trow[idx] = (h16)hv;
        }
      }
      __syncthreads();
    }
  }

  // ---- stage 3: MHA over 8 tokens ----
  h16* QKV = SCR;        // 16x192
  h16* OT = SCR + 3072;  // 16x64
  const h16* WQ = W + 49152;
  const h16* WOp = W + 61440;
  for (int mt = 0; mt < 8; ++mt) {  // 16 token-rows (2 samples) per tile
    v16h a0 = ldsA(TOK + mt * 16 * 64, 64, r, 0, koff);
    v16h a1 = ldsA(TOK + mt * 16 * 64, 64, r, 32, koff);
#pragma unroll 4
    for (int nt = 0; nt < 12; ++nt) {
      v8f c = zero8();
      c = wmma16(a0, ldsB(WQ, 64, nt * 16 + r, 0, ksel), c);
      c = wmma16(a1, ldsB(WQ, 64, nt * 16 + r, 32, ksel), c);
      int col = nt * 16 + r;
      float bq = M[2624 + col];
#pragma unroll
      for (int v = 0; v < 8; ++v) QKV[(v + hf * 8) * 192 + col] = (h16)(c[v] + bq);
    }
    __syncthreads();
    {  // per-lane attention: lane half picks 4 of 8 heads; 8x8, dh=8 -> too small for WMMA tiles
      int kb = (r >> 3) * 8;
      for (int h = hf * 4; h < hf * 4 + 4; ++h) {
        float q[8];
#pragma unroll
        for (int d0 = 0; d0 < 8; ++d0) q[d0] = (float)QKV[r * 192 + h * 8 + d0];
        float sc[8]; float mx = -1e30f;
#pragma unroll
        for (int t = 0; t < 8; ++t) {
          float s = 0.f;
#pragma unroll
          for (int d0 = 0; d0 < 8; ++d0) s += q[d0] * (float)QKV[(kb + t) * 192 + 64 + h * 8 + d0];
          sc[t] = s * 0.35355339059f;
          mx = fmaxf(mx, sc[t]);
        }
        float se = 0.f;
#pragma unroll
        for (int t = 0; t < 8; ++t) { sc[t] = __expf(sc[t] - mx); se += sc[t]; }
        float inv = 1.f / se;
#pragma unroll
        for (int d0 = 0; d0 < 8; ++d0) {
          float o = 0.f;
#pragma unroll
          for (int t = 0; t < 8; ++t) o += sc[t] * (float)QKV[(kb + t) * 192 + 128 + h * 8 + d0];
          OT[r * 64 + h * 8 + d0] = (h16)(o * inv);
        }
      }
    }
    __syncthreads();
    {  // output projection + softmax-weighted token fusion
      v16h o0 = ldsA(OT, 64, r, 0, koff);
      v16h o1 = ldsA(OT, 64, r, 32, koff);
      int sw = mt * 2 + hf;  // sample within wave
#pragma unroll
      for (int nt = 0; nt < 4; ++nt) {
        v8f c = zero8();
        c = wmma16(o0, ldsB(WOp, 64, nt * 16 + r, 0, ksel), c);
        c = wmma16(o1, ldsB(WOp, 64, nt * 16 + r, 32, ksel), c);
        int col = nt * 16 + r;
        float bov = M[2816 + col];
        float s0 = 0.f, s1 = 0.f;
#pragma unroll
        for (int v = 0; v < 4; ++v) {
          s0 += M[2880 + v] * (c[v] + bov);       // tokens 0..3 -> f1
          s1 += M[2880 + v] * (c[4 + v] + bov);   // tokens 4..7 -> f2
        }
        FB[sw * 128 + col] = (h16)s0;
        FB[sw * 128 + 64 + col] = (h16)s1;
      }
    }
    __syncthreads();
  }

  // ---- stage 4: classifier ----
  h16* CH = SCR;  // 16x64
  const h16* WC1 = W + 65536;
#pragma unroll
  for (int nt = 0; nt < 4; ++nt) {
    v8f c = zero8();
#pragma unroll
    for (int kc = 0; kc < 4; ++kc)
      c = wmma16(ldsA(FB, 128, r, kc * 32, koff), ldsB(WC1, 128, nt * 16 + r, kc * 32, ksel), c);
    int col = nt * 16 + r;
    float bv = M[2884 + col];
#pragma unroll
    for (int v = 0; v < 8; ++v) CH[(v + hf * 8) * 64 + col] = (h16)(c[v] + bv);
  }
  __syncthreads();
  {  // distributed LN + relu + 64->1 dot (all 32 lanes; FB region now dead -> exchange)
    float s = 0.f, q = 0.f;
#pragma unroll
    for (int cc = 0; cc < 32; ++cc) {
      float v0 = (float)CH[r * 64 + hf * 32 + cc];
      s += v0; q += v0 * v0;
    }
    EXF[lane] = s; EXF[64 + lane] = q;
    __syncthreads();
    float st = EXF[r] + EXF[r + 16];
    float qt = EXF[64 + r] + EXF[64 + r + 16];
    float mu = st * (1.f / 64.f);
    float var = qt * (1.f / 64.f) - mu * mu;
    float inv = rsqrtf(var + 1e-5f);
    float dp = 0.f;
#pragma unroll
    for (int cc = 0; cc < 32; ++cc) {
      int idx = hf * 32 + cc;
      float hh = ((float)CH[r * 64 + idx] - mu) * inv * M[2948 + idx] + M[3012 + idx];
      hh = fmaxf(hh, 0.f);
      dp += hh * M[3076 + idx];
    }
    EXF[128 + lane] = dp;
    __syncthreads();
    if (lane < 16) out[base + r] = EXF[128 + r] + EXF[128 + r + 16] + M[3140];
  }
}

// =================== versus-loss reduction ===================
__global__ void __launch_bounds__(256) ver_reduce(const float* __restrict__ part, int nblk,
                                                  float* __restrict__ out, int N) {
  __shared__ float sh[256];
  float s = 0.f;
  for (int i = threadIdx.x; i < nblk; i += 256) s += part[i];
  sh[threadIdx.x] = s;
  __syncthreads();
  if (threadIdx.x == 0) {
    float t = 0.f;
    for (int i = 0; i < 256; ++i) t += sh[i];
    out[N] = t / (float)N;
  }
}

extern "C" void kernel_launch(void* const* d_in, const int* in_sizes, int n_in,
                              void* d_out, int out_size, void* d_ws, size_t ws_size,
                              hipStream_t stream) {
  (void)out_size; (void)ws_size;
  Params P;
  for (int i = 0; i < NPAR; ++i) P.a[i] = (const float*)d_in[i < n_in ? i : 0];
  const int N = in_sizes[0] / 16;
  h16* wf16 = (h16*)d_ws;
  float* misc = (float*)((char*)d_ws + 147456);
  float* partials = (float*)((char*)d_ws + 163840);

  pack_weights<<<64, 256, 0, stream>>>(P, wf16, misc);

  const int blocks = N / 64;  // 64 samples per block (4 waves x 16)
  const size_t shmem = 147456 + 13312 + 4 * 34816;  // 300032 B < 320 KB/WGP
  hipFuncSetAttribute((const void*)hydranet_fused,
                      hipFuncAttributeMaxDynamicSharedMemorySize, (int)shmem);
  hydranet_fused<<<blocks, 128, shmem, stream>>>((const float*)d_in[0], (const float*)d_in[1],
                                                 wf16, misc, (float*)d_out, partials);
  ver_reduce<<<1, 256, 0, stream>>>(partials, blocks, (float*)d_out, N);
}
